// spatial_attention_16750372455187
// MI455X (gfx1250) — compile-verified, matrix operand order checked by reading, not run on hardware
//
#include <hip/hip_runtime.h>
#include <math.h>

// N = 4096 nodes, F = 128 features, E = 64 embed
// out = concat([weight@t.T, weight@(wf@t).T, weight@(wb.T@t).T, weight@(wg@t).T]) : (256, 4096)

#define NN 4096
#define FF 128

typedef __attribute__((ext_vector_type(16))) __bf16 v16bf;
typedef __attribute__((ext_vector_type(8)))  float  v8f;

#if __has_builtin(__builtin_amdgcn_global_load_async_to_lds_b128) && \
    __has_builtin(__builtin_amdgcn_s_wait_asynccnt)
#define USE_ASYNC_LDS 1
typedef int async_v4i __attribute__((vector_size(4 * sizeof(int))));
typedef __attribute__((address_space(1))) async_v4i* gas_ptr;
typedef __attribute__((address_space(3))) async_v4i* las_ptr;
#else
#define USE_ASYNC_LDS 0
#endif

// ---------------------------------------------------------------------------
// Kernel 1: v1_x = W_x @ a1_x, v2_x = W_x @ a2_x   (six F-vectors)
// ---------------------------------------------------------------------------
__global__ void k_vvec(const float* __restrict__ Wf, const float* __restrict__ af1, const float* __restrict__ af2,
                       const float* __restrict__ Wb, const float* __restrict__ ab1, const float* __restrict__ ab2,
                       const float* __restrict__ Wg, const float* __restrict__ ag1, const float* __restrict__ ag2,
                       float* __restrict__ V) {
    int f = threadIdx.x; // 0..127
    const float* Ws[3] = {Wf, Wb, Wg};
    const float* A1[3] = {af1, ab1, ag1};
    const float* A2[3] = {af2, ab2, ag2};
    for (int x = 0; x < 3; ++x) {
        float s1 = 0.f, s2 = 0.f;
        for (int e = 0; e < 64; ++e) {
            float w = Ws[x][f * 64 + e];
            s1 += w * A1[x][e];
            s2 += w * A2[x][e];
        }
        V[(2 * x) * FF + f]     = s1;  // v1_x
        V[(2 * x + 1) * FF + f] = s2;  // v2_x
    }
}

// ---------------------------------------------------------------------------
// Kernel 2: p1_x[n] = feat[n,:] . v1_x ; p2_x[n] = feat[n,:] . v2_x
// ---------------------------------------------------------------------------
__global__ void k_pvec(const float* __restrict__ feat, const float* __restrict__ V,
                       float* __restrict__ P1, float* __restrict__ P2) {
    int node = blockIdx.x;
    int f = threadIdx.x; // 128 threads
    float xv = feat[(size_t)node * FF + f];
    __shared__ float sm[6][128];
    for (int c = 0; c < 6; ++c) sm[c][f] = xv * V[c * FF + f];
    __syncthreads();
    for (int s = 64; s > 0; s >>= 1) {
        if (f < s)
            for (int c = 0; c < 6; ++c) sm[c][f] += sm[c][f + s];
        __syncthreads();
    }
    if (f < 6) {
        int x = f >> 1;
        float v = sm[f][0];
        if ((f & 1) == 0) P1[x * NN + node] = v;
        else              P2[x * NN + node] = v;
    }
}

// ---------------------------------------------------------------------------
// Kernel 2b: featT[f][n] = (bf16) feat[n][f]   (pre-transposed bf16 features,
//            1 MB, built once so the hot GEMM loop stages B with pure 16B copies)
// ---------------------------------------------------------------------------
__global__ void __launch_bounds__(256) k_featT(const float* __restrict__ feat,
                                               unsigned int* __restrict__ fT32) {
    __shared__ float S[32][129];
    int n0 = blockIdx.x * 32;
    int tid = threadIdx.x;
    for (int c = 0; c < 16; ++c) {
        int idx = c * 256 + tid;        // 0..4095
        int nd = idx >> 7, f = idx & 127;
        S[nd][f] = feat[(size_t)(n0 + nd) * FF + f];
    }
    __syncthreads();
    for (int c = 0; c < 8; ++c) {
        int idx = c * 256 + tid;        // 0..2047
        int np = idx & 15, f = idx >> 4;
        unsigned int u0 = __builtin_bit_cast(unsigned short, (__bf16)S[2 * np][f]);
        unsigned int u1 = __builtin_bit_cast(unsigned short, (__bf16)S[2 * np + 1][f]);
        fT32[(size_t)f * (NN / 2) + (n0 >> 1) + np] = u0 | (u1 << 16);
    }
}

// ---------------------------------------------------------------------------
// Kernel 3: s2_x[i] = (sum_j adj_ij * p2_x[j]) / max(deg_i, 1)
// ---------------------------------------------------------------------------
__global__ void k_s2(const float* __restrict__ adjF, const float* __restrict__ adjB,
                     const float* __restrict__ adjG,
                     const float* __restrict__ P2, float* __restrict__ S2) {
    int x = blockIdx.y;
    const float* adj = (x == 0) ? adjF : (x == 1) ? adjB : adjG;
    const float* p2 = P2 + x * NN;
    int i = blockIdx.x, tid = threadIdx.x;
    float deg = 0.f, q = 0.f;
    for (int j = tid; j < NN; j += 256) {
        float a = adj[(size_t)i * NN + j];
        if (a > 0.f) { deg += a; q += a * p2[j]; }
    }
    __shared__ float sd[256], sq[256];
    sd[tid] = deg; sq[tid] = q;
    __syncthreads();
    for (int s = 128; s > 0; s >>= 1) {
        if (tid < s) { sd[tid] += sd[tid + s]; sq[tid] += sq[tid + s]; }
        __syncthreads();
    }
    if (tid == 0) {
        float d = sd[0] < 1.f ? 1.f : sd[0];
        S2[x * NN + i] = sq[0] / d;
    }
}

// ---------------------------------------------------------------------------
// Kernel 4: per-row online-softmax stats  m_i, 1/Z_i  over neighbors
//           e_ij = leakyrelu(p1_i + s2_j, 0.2)
// ---------------------------------------------------------------------------
__global__ void k_mz(const float* __restrict__ adjF, const float* __restrict__ adjB,
                     const float* __restrict__ adjG,
                     const float* __restrict__ P1, const float* __restrict__ S2,
                     float* __restrict__ Mx, float* __restrict__ IZ) {
    int x = blockIdx.y;
    const float* adj = (x == 0) ? adjF : (x == 1) ? adjB : adjG;
    const float* p1 = P1 + x * NN;
    const float* s2 = S2 + x * NN;
    int i = blockIdx.x, tid = threadIdx.x;
    float p1i = p1[i];
    float ml = -3.0e38f, zl = 0.f;
    for (int j = tid; j < NN; j += 256) {
        float a = adj[(size_t)i * NN + j];
        if (a > 0.f) {
            float v = p1i + s2[j];
            v = v > 0.f ? v : 0.2f * v;
            if (v > ml) { zl *= __expf(ml - v); ml = v; }
            zl += __expf(v - ml);
        }
    }
    __shared__ float sm[256], sz[256];
    sm[tid] = ml; sz[tid] = zl;
    __syncthreads();
    for (int s = 128; s > 0; s >>= 1) {
        if (tid < s) {
            float m1 = sm[tid], z1 = sz[tid];
            float m2 = sm[tid + s], z2 = sz[tid + s];
            float mmx = fmaxf(m1, m2);
            sm[tid] = mmx;
            sz[tid] = z1 * __expf(m1 - mmx) + z2 * __expf(m2 - mmx);
        }
        __syncthreads();
    }
    if (tid == 0) {
        Mx[x * NN + i] = sm[0];
        IZ[x * NN + i] = 1.0f / sz[0];
    }
}

// ---------------------------------------------------------------------------
// Kernel 5: WMMA aggregation  agg_x = P_x @ t   (P built on the fly from adj)
//   x==1 (backward) uses transposed adjacency with its-row stats.
//   Block: 256 threads (8 waves); 16 output rows (i) per block, 16-wide f
//   slice per wave. K (=j) staged 128 at a time.
//   B tile comes from pre-transposed bf16 featT -> staged with async
//   global->LDS b128 copies (ASYNCcnt) when available.  Both A and B tiles
//   are laid out so each lane's fragment is two contiguous 16B runs ->
//   ds_load_b128 pairs feed each v_wmma_f32_16x16x32_bf16.
// ---------------------------------------------------------------------------
#define LROW 68            // padded row stride in dwords (136 bf16)

__global__ void __launch_bounds__(256) k_agg(
    const float* __restrict__ adjF, const float* __restrict__ adjB, const float* __restrict__ adjG,
    const __bf16* __restrict__ featT,
    const float* __restrict__ P1, const float* __restrict__ S2,
    const float* __restrict__ Mx, const float* __restrict__ IZ,
    float* __restrict__ AGG) {
    int x = blockIdx.y;
    const float* adj = (x == 0) ? adjF : (x == 1) ? adjB : adjG;
    const int mode = (x == 1) ? 1 : 0;  // 1 => transposed access (w_b.T @ t)
    const float* p1 = P1 + x * NN;
    const float* s2 = S2 + x * NN;
    const float* mm = Mx + x * NN;
    const float* iz = IZ + x * NN;
    float* agg = AGG + (size_t)x * NN * FF;

    __shared__ unsigned int Pl32[16 * LROW];    // P tile: 16 rows(i) x 128 k(j), packed bf16 pairs
    __shared__ unsigned int Tl32[128 * LROW];   // t tile TRANSPOSED: 128 rows(f) x 128 k(j)
    const __bf16* Pl = (const __bf16*)Pl32;
    const __bf16* Tl = (const __bf16*)Tl32;

    const int tid  = threadIdx.x;
    const int lane = tid & 31;
    const int wave = tid >> 5;      // 0..7
    const int f0   = wave * 16;
    const int i0   = blockIdx.x * 16;

    v8f acc = {};

    for (int jc = 0; jc < NN; jc += 128) {
        // ---- stage t tile (f-major) from featT: 2048 x 16B block copies ----
#if USE_ASYNC_LDS
        #pragma unroll
        for (int c = 0; c < 8; ++c) {
            int idx = c * 256 + tid;           // 0..2047
            int f = idx >> 4, cw = idx & 15;   // 16B chunk cw of row f
            const __bf16* g = featT + (size_t)f * NN + jc + cw * 8;
            __builtin_amdgcn_global_load_async_to_lds_b128(
                (gas_ptr)(void*)g, (las_ptr)(void*)&Tl32[f * LROW + cw * 4], 0, 0);
        }
#else
        #pragma unroll
        for (int c = 0; c < 8; ++c) {
            int idx = c * 256 + tid;
            int f = idx >> 4, cw = idx & 15;
            const uint4* g = (const uint4*)(featT + (size_t)f * NN + jc) + cw;
            *(uint4*)&Tl32[f * LROW + cw * 4] = *g;
        }
#endif
        // ---- build softmax-weight tile P (bf16, packed pairs along j) ----
        if (mode == 0) {
            #pragma unroll
            for (int c = 0; c < 4; ++c) {
                int p = c * 256 + tid;         // 0..1023
                int row = p >> 6, cp = p & 63; // pairs along j, coalesced
                int j = jc + 2 * cp;
                float a0 = adj[(size_t)(i0 + row) * NN + j];
                float a1 = adj[(size_t)(i0 + row) * NN + j + 1];
                if (jc + 128 < NN)
                    __builtin_prefetch(&adj[(size_t)(i0 + row) * NN + j + 128], 0, 3);
                float pi = p1[i0 + row], mi = mm[i0 + row], zi = iz[i0 + row];
                float w0 = 0.f, w1 = 0.f;
                if (a0 > 0.f) { float v = pi + s2[j];     v = v > 0.f ? v : 0.2f * v; w0 = __expf(v - mi) * zi; }
                if (a1 > 0.f) { float v = pi + s2[j + 1]; v = v > 0.f ? v : 0.2f * v; w1 = __expf(v - mi) * zi; }
                unsigned int u0 = __builtin_bit_cast(unsigned short, (__bf16)w0);
                unsigned int u1 = __builtin_bit_cast(unsigned short, (__bf16)w1);
                Pl32[row * LROW + cp] = u0 | (u1 << 16);
            }
        } else {
            #pragma unroll
            for (int c = 0; c < 4; ++c) {
                int p = c * 256 + tid;
                int row = p & 15, cp = p >> 4; // 16 consecutive i per 16 lanes
                int j = jc + 2 * cp;
                float a0 = adj[(size_t)j       * NN + (i0 + row)];
                float a1 = adj[(size_t)(j + 1) * NN + (i0 + row)];
                if (jc + 128 < NN)
                    __builtin_prefetch(&adj[(size_t)(j + 128) * NN + (i0 + row)], 0, 3);
                float si = s2[i0 + row];
                float w0 = 0.f, w1 = 0.f;
                if (a0 > 0.f) { float v = p1[j]     + si; v = v > 0.f ? v : 0.2f * v; w0 = __expf(v - mm[j])     * iz[j];     }
                if (a1 > 0.f) { float v = p1[j + 1] + si; v = v > 0.f ? v : 0.2f * v; w1 = __expf(v - mm[j + 1]) * iz[j + 1]; }
                unsigned int u0 = __builtin_bit_cast(unsigned short, (__bf16)w0);
                unsigned int u1 = __builtin_bit_cast(unsigned short, (__bf16)w1);
                Pl32[row * LROW + cp] = u0 | (u1 << 16);
            }
        }
#if USE_ASYNC_LDS
        __builtin_amdgcn_s_wait_asynccnt(0);   // own async LDS writes landed
#endif
        __syncthreads();

        // ---- 4 WMMA k-substeps of 32; fragments = 2 contiguous 16B runs ----
        const int rowA = lane & 15;
        const int kbase = (lane < 16) ? 0 : 8;
        const int colB = f0 + rowA;
        #pragma unroll
        for (int ks = 0; ks < 4; ++ks) {
            int kb = ks * 32;
            v16bf afrag, bfrag;
            #pragma unroll
            for (int v = 0; v < 8; ++v) {
                int k = kb + kbase + ((v < 4) ? (2 * v) : (16 + 2 * (v - 4)));
                afrag[2 * v]     = Pl[rowA * (2 * LROW) + k];
                afrag[2 * v + 1] = Pl[rowA * (2 * LROW) + k + 1];
                bfrag[2 * v]     = Tl[colB * (2 * LROW) + k];
                bfrag[2 * v + 1] = Tl[colB * (2 * LROW) + k + 1];
            }
            acc = __builtin_amdgcn_wmma_f32_16x16x32_bf16(
                false, afrag, false, bfrag, (short)0, acc, false, false);
        }
        __syncthreads();
    }

    // ---- store C (16x16 f32 tile): VGPR r -> M = r + (lane>=16 ? 8 : 0) ----
    const int base = (lane < 16) ? 0 : 8;
    const int colC = lane & 15;
    #pragma unroll
    for (int r = 0; r < 8; ++r) {
        agg[(size_t)(i0 + base + r) * FF + f0 + colC] = acc[r];
    }
}

// ---------------------------------------------------------------------------
// Kernel 6: out[r, n] = sum_f weight[r%64, f] * src_{r/64}[n, f]
//   src: 0=features, 1=agg_f, 2=agg_b, 3=agg_g.  Block handles 16 nodes.
// ---------------------------------------------------------------------------
__global__ void __launch_bounds__(256) k_proj(const float* __restrict__ weight,
                                              const float* __restrict__ feat,
                                              const float* __restrict__ AGG,
                                              float* __restrict__ out) {
    __shared__ float S[4][16][129];   // padded against bank conflicts
    int n0 = blockIdx.x * 16;
    int tid = threadIdx.x;
    for (int c = 0; c < 32; ++c) {
        int idx = c * 256 + tid;      // 0..8191
        int g = idx >> 11;
        int rem = idx & 2047;
        int nd = rem >> 7, f = rem & 127;
        const float* src = (g == 0) ? feat : (AGG + (size_t)(g - 1) * NN * FF);
        S[g][nd][f] = src[(size_t)(n0 + nd) * FF + f];
    }
    __syncthreads();
    int n  = tid & 15;
    int rc = tid >> 4;                // 0..15
    for (int rr = 0; rr < 16; ++rr) {
        int r = rc * 16 + rr;         // 0..255
        int g = r >> 6, e = r & 63;
        float s = 0.f;
        for (int f = 0; f < FF; ++f) s += weight[e * FF + f] * S[g][n][f];
        out[(size_t)r * NN + n0 + n] = s;
    }
}

// ---------------------------------------------------------------------------
extern "C" void kernel_launch(void* const* d_in, const int* in_sizes, int n_in,
                              void* d_out, int out_size, void* d_ws, size_t ws_size,
                              hipStream_t stream) {
    (void)in_sizes; (void)n_in; (void)out_size; (void)ws_size;
    const float* feat  = (const float*)d_in[0];
    const float* adjG  = (const float*)d_in[1];
    const float* adjF  = (const float*)d_in[2];
    const float* adjB  = (const float*)d_in[3];
    const float* weight= (const float*)d_in[4];
    const float* Wf    = (const float*)d_in[5];
    const float* af1   = (const float*)d_in[6];
    const float* af2   = (const float*)d_in[7];
    const float* Wb    = (const float*)d_in[8];
    const float* ab1   = (const float*)d_in[9];
    const float* ab2   = (const float*)d_in[10];
    const float* Wg    = (const float*)d_in[11];
    const float* ag1   = (const float*)d_in[12];
    const float* ag2   = (const float*)d_in[13];

    float* ws = (float*)d_ws;
    float* V     = ws;                 //   768 floats
    float* P1    = V  + 768;           // 3*4096
    float* P2    = P1 + 3 * NN;
    float* S2    = P2 + 3 * NN;
    float* Mx    = S2 + 3 * NN;
    float* IZ    = Mx + 3 * NN;
    float* AGG   = IZ + 3 * NN;        // 3*4096*128 floats (6 MB)
    float* featTf= AGG + 3 * NN * FF;  // 128*4096 bf16 (1 MB), 16B-aligned offset
    __bf16* featT = (__bf16*)featTf;
    float* out = (float*)d_out;

    k_vvec <<<1, 128, 0, stream>>>(Wf, af1, af2, Wb, ab1, ab2, Wg, ag1, ag2, V);
    k_pvec <<<NN, 128, 0, stream>>>(feat, V, P1, P2);
    k_featT<<<NN / 32, 256, 0, stream>>>(feat, (unsigned int*)featT);
    k_s2 <<<dim3(NN, 3), 256, 0, stream>>>(adjF, adjB, adjG, P2, S2);
    k_mz <<<dim3(NN, 3), 256, 0, stream>>>(adjF, adjB, adjG, P1, S2, Mx, IZ);
    k_agg<<<dim3(NN / 16, 3), 256, 0, stream>>>(adjF, adjB, adjG, featT, P1, S2, Mx, IZ, AGG);
    k_proj<<<NN / 16, 256, 0, stream>>>(weight, feat, AGG, out);
}
